// HybridGCN_SAGE_39238821216992
// MI455X (gfx1250) — matrix operand order repaired and checked by reading, not compiled, not run on hardware
//
#include <hip/hip_runtime.h>

typedef __attribute__((ext_vector_type(16))) _Float16 v16h;
typedef __attribute__((ext_vector_type(8)))  _Float16 v8h;
typedef __attribute__((ext_vector_type(8)))  float    v8f;
typedef unsigned int u32;
typedef __attribute__((ext_vector_type(4))) u32 u32x4;
typedef __attribute__((ext_vector_type(8))) int i32x8;
typedef __attribute__((ext_vector_type(4))) int i32x4;

#define HID 128
#define OUTD 3

// ---------------------------------------------------------------------------
// Tensor Data Mover: 1-D contiguous copy of `nelem` f16 elements into LDS.
// D# layout per CDNA5 ISA:
//  group0: [1:0]=count=1, [63:32]=lds_addr, [120:64]=global_addr, [127:126]=type=2
//  group1: [17:16]=data_size=1 (2B), [79:48]=tensor_dim0, [127:112]=tile_dim0,
//          [207:160]=tensor_dim0_stride ; everything else 0 (1-D tile)
//  groups 2/3: zero (<=2D tensor). 6-arg builtin form (clang-23 toolchain).
// ---------------------------------------------------------------------------
__device__ inline u32 lds_off(const void* p) {
  // LDS aperture: LDS offset is the low 32 bits of the generic address
  return (u32)(size_t)p;
}

__device__ inline void tdm_load_1d_f16(const void* gsrc, u32 lds_byte_off, u32 nelem) {
  unsigned long long ga = (unsigned long long)(size_t)gsrc;
  u32x4 g0;
  g0[0] = 1u;                                        // count=1, user mode
  g0[1] = lds_byte_off;                              // lds_addr
  g0[2] = (u32)(ga & 0xFFFFFFFFu);                   // global_addr[31:0]
  g0[3] = (u32)((ga >> 32) & 0x01FFFFFFu) | (2u << 30); // addr[56:32] | type=2
  i32x8 g1;
  g1[0] = (int)(1u << 16);                           // data_size=1 (2 bytes)
  g1[1] = (int)((nelem & 0xFFFFu) << 16);            // tensor_dim0[15:0]
  g1[2] = (int)((nelem >> 16) & 0xFFFFu);            // tensor_dim0[31:16]
  g1[3] = (int)((nelem & 0xFFFFu) << 16);            // tile_dim0 (<= 16384 here)
  g1[4] = 0;                                         // tile_dim1/2 unused
  g1[5] = (int)nelem;                                // tensor_dim0_stride[31:0]
  g1[6] = 0;
  g1[7] = 0;
  i32x4 gz4 = {0, 0, 0, 0};
  i32x8 gz8 = {0, 0, 0, 0, 0, 0, 0, 0};
  __builtin_amdgcn_tensor_load_to_lds(g0, g1, gz4, gz4, gz8, 0);
}

// ---------------------------------------------------------------------------
// WMMA fragment gathers from LDS (layouts per ISA 7.12.2, wave32).
// ---------------------------------------------------------------------------
__device__ inline v16h cat8(v8h lo, v8h hi) {
  v16h r;
#pragma unroll
  for (int i = 0; i < 8; ++i) { r[i] = lo[i]; r[8 + i] = hi[i]; }
  return r;
}

// A 16x32 (MxK) frag from LDS tile sA[16][128] (row-major f16).
// lane 0-15 : M=lane,    K halves {k0+0..7,  k0+16..23}
// lane 16-31: M=lane-16, K halves {k0+8..15, k0+24..31}
__device__ inline v16h lds_a_frag(const _Float16* sA, int k0, int lane) {
  const _Float16* p = sA + (lane & 15) * HID + k0 + (((lane >> 4) << 3));
  v8h lo = *(const v8h*)(p);
  v8h hi = *(const v8h*)(p + 16);
  return cat8(lo, hi);
}

// B 32x16 (KxN) frag from LDS transposed weight sW[n][k] (f16, 128x128).
// lane 0-15 : N=col0+lane, K={k0+0..7, k0+16..23} ; lane 16-31: N=col0+lane-16, K={k0+8..15, k0+24..31}
__device__ inline v16h lds_b_frag(const _Float16* sW, int col0, int k0, int lane) {
  const _Float16* p = sW + (col0 + (lane & 15)) * HID + k0 + ((lane >> 4) << 3);
  v8h lo = *(const v8h*)(p);
  v8h hi = *(const v8h*)(p + 16);
  return cat8(lo, hi);
}

// ---------------------------------------------------------------------------
// OUT[M x 128] = A1h @ W1t^T (+ A2h @ W2t^T) (+bias)(relu). A* are f16 row-major
// [M][128]; W*t are f16 transposed weights [n][k]. grid.x = M/16; block = 256.
// Operands staged into LDS by the Tensor Data Mover (wave 0), consumed by all
// 8 waves; wave w computes output column tile w with 4 (or 8) v_wmma ops.
// ---------------------------------------------------------------------------
__global__ void __launch_bounds__(256)
gemm128_f16(const _Float16* __restrict__ A1, const _Float16* __restrict__ W1t,
            const _Float16* __restrict__ A2, const _Float16* __restrict__ W2t,
            const float* __restrict__ bias, float* __restrict__ OUT, int do_relu) {
  __shared__ __align__(16) _Float16 sW1[HID * HID];
  __shared__ __align__(16) _Float16 sW2[HID * HID];
  __shared__ __align__(16) _Float16 sA1[16 * HID];
  __shared__ __align__(16) _Float16 sA2[16 * HID];
  const int lane = threadIdx.x & 31;
  const int row0 = blockIdx.x << 4;
  const int col0 = (threadIdx.x >> 5) << 4;
  const bool dual = (A2 != nullptr);

  if (threadIdx.x < 32) {   // wave 0 issues the async tensor DMAs
    tdm_load_1d_f16(W1t, lds_off(sW1), HID * HID);
    tdm_load_1d_f16(A1 + (size_t)row0 * HID, lds_off(sA1), 16 * HID);
    if (dual) {
      tdm_load_1d_f16(W2t, lds_off(sW2), HID * HID);
      tdm_load_1d_f16(A2 + (size_t)row0 * HID, lds_off(sA2), 16 * HID);
    }
    __builtin_amdgcn_s_wait_tensorcnt(0);
  }
  __syncthreads();

  v8f c = {};
#pragma unroll
  for (int k0 = 0; k0 < HID; k0 += 32) {
    v16h a = lds_a_frag(sA1, k0, lane);
    v16h b = lds_b_frag(sW1, col0, k0, lane);
    c = __builtin_amdgcn_wmma_f32_16x16x32_f16(false, a, false, b,
                                               (short)0, c, false, false);
  }
  if (dual) {
#pragma unroll
    for (int k0 = 0; k0 < HID; k0 += 32) {
      v16h a = lds_a_frag(sA2, k0, lane);
      v16h b = lds_b_frag(sW2, col0, k0, lane);
      c = __builtin_amdgcn_wmma_f32_16x16x32_f16(false, a, false, b,
                                                 (short)0, c, false, false);
    }
  }
  // C/D layout: lane L, elem v -> M = row0 + v + (L>>4)*8, N = col0 + (L&15)
  const int cn = col0 + (lane & 15);
  const int rb = row0 + ((lane >> 4) << 3);
  const float bv = (bias != nullptr) ? bias[cn] : 0.0f;
#pragma unroll
  for (int v = 0; v < 8; ++v) {
    float val = c[v] + bv;
    if (do_relu) val = fmaxf(val, 0.0f);
    OUT[(size_t)(rb + v) * HID + cn] = val;
  }
}

// ---------------- conversion kernels ----------------------------------------
// Wt[n*128+k] = (f16) W[k*128+n]
__global__ void wt_convert(const float* __restrict__ W, _Float16* __restrict__ Wt) {
  int idx = blockIdx.x * blockDim.x + threadIdx.x;
  if (idx >= HID * HID) return;
  int n = idx >> 7, k = idx & (HID - 1);
  Wt[idx] = (_Float16)W[k * HID + n];
}

__global__ void f32_to_f16(const float* __restrict__ in, _Float16* __restrict__ out,
                           int n) {
  int i = blockIdx.x * blockDim.x + threadIdx.x;
  if (i < n) out[i] = (_Float16)in[i];
}

// ---------------- graph / elementwise kernels --------------------------------
__global__ void fill_f32(float* __restrict__ p, int n, float v) {
  int i = blockIdx.x * blockDim.x + threadIdx.x;
  if (i < n) p[i] = v;
}

__global__ void deg_scatter(const int* __restrict__ col, float* __restrict__ deg, int E) {
  int e = blockIdx.x * blockDim.x + threadIdx.x;
  if (e < E) atomicAdd(&deg[col[e]], 1.0f);
}

__global__ void dinv_kernel(const float* __restrict__ deg, float* __restrict__ dinv, int n) {
  int i = blockIdx.x * blockDim.x + threadIdx.x;
  if (i < n) {
    float d = deg[i];
    dinv[i] = (d > 0.0f) ? rsqrtf(d) : 0.0f;
  }
}

__global__ void agg_init(const float* __restrict__ h, const float* __restrict__ dinv,
                         float* __restrict__ agg, int total) {
  int idx = blockIdx.x * blockDim.x + threadIdx.x;
  if (idx >= total) return;
  float di = dinv[idx >> 7];
  agg[idx] = h[idx] * di * di;
}

__global__ void gcn_edge_scatter(const float* __restrict__ h, const float* __restrict__ dinv,
                                 const int* __restrict__ row, const int* __restrict__ col,
                                 float* __restrict__ agg, int E) {
  int e = blockIdx.x * (blockDim.x >> 5) + (threadIdx.x >> 5);
  if (e >= E) return;
  const int lane = threadIdx.x & 31;
  const int r = row[e], c = col[e];
  const float nrm = dinv[r] * dinv[c];
  const float4 v = *(const float4*)(h + (size_t)r * HID + lane * 4);
  float* op = agg + (size_t)c * HID + lane * 4;
  atomicAdd(op + 0, v.x * nrm);
  atomicAdd(op + 1, v.y * nrm);
  atomicAdd(op + 2, v.z * nrm);
  atomicAdd(op + 3, v.w * nrm);
}

__global__ void bias_relu_inplace(float* __restrict__ a, const float* __restrict__ bias,
                                  int total) {
  int idx = blockIdx.x * blockDim.x + threadIdx.x;
  if (idx >= total) return;
  a[idx] = fmaxf(a[idx] + bias[idx & (HID - 1)], 0.0f);
}

__global__ void sage_edge_scatter(const float* __restrict__ hg,
                                  const int* __restrict__ row, const int* __restrict__ col,
                                  float* __restrict__ neigh, float* __restrict__ cnt, int E) {
  int e = blockIdx.x * (blockDim.x >> 5) + (threadIdx.x >> 5);
  if (e >= E) return;
  const int lane = threadIdx.x & 31;
  const int r = row[e], c = col[e];
  const float4 v = *(const float4*)(hg + (size_t)r * HID + lane * 4);
  float* op = neigh + (size_t)c * HID + lane * 4;
  atomicAdd(op + 0, v.x);
  atomicAdd(op + 1, v.y);
  atomicAdd(op + 2, v.z);
  atomicAdd(op + 3, v.w);
  if (lane == 0) atomicAdd(&cnt[c], 1.0f);
}

__global__ void sage_norm(float* __restrict__ neigh, const float* __restrict__ cnt,
                          int total) {
  int idx = blockIdx.x * blockDim.x + threadIdx.x;
  if (idx >= total) return;
  neigh[idx] = neigh[idx] / fmaxf(cnt[idx >> 7], 1.0f);
}

__global__ void pool_scatter(const float* __restrict__ hs, const int* __restrict__ batch,
                             float* __restrict__ pooled, int total) {
  int idx = blockIdx.x * blockDim.x + threadIdx.x;
  if (idx >= total) return;
  const int node = idx >> 7;
  const int d = idx & (HID - 1);
  atomicAdd(&pooled[(size_t)batch[node] * HID + d], hs[idx]);
}

__global__ void fc2_kernel(const float* __restrict__ z, const float* __restrict__ W,
                           const float* __restrict__ b, float* __restrict__ out, int G) {
  int t = blockIdx.x * blockDim.x + threadIdx.x;
  if (t >= G * OUTD) return;
  const int g = t / OUTD, o = t % OUTD;
  float s = b[o];
  const float* zp = z + (size_t)g * HID;
#pragma unroll 8
  for (int k = 0; k < HID; ++k) s += zp[k] * W[k * OUTD + o];
  out[t] = s;
}

// ---------------- launch -----------------------------------------------------
extern "C" void kernel_launch(void* const* d_in, const int* in_sizes, int n_in,
                              void* d_out, int out_size, void* d_ws, size_t ws_size,
                              hipStream_t stream) {
  const float* x        = (const float*)d_in[0];
  const int*   ei       = (const int*)  d_in[1];
  const int*   batch    = (const int*)  d_in[2];
  const float* W_gcn    = (const float*)d_in[3];
  const float* b_gcn    = (const float*)d_in[4];
  const float* W_sage_l = (const float*)d_in[5];
  const float* W_sage_r = (const float*)d_in[6];
  const float* b_sage   = (const float*)d_in[7];
  const float* W_fc1    = (const float*)d_in[8];
  const float* b_fc1    = (const float*)d_in[9];
  const float* W_fc2    = (const float*)d_in[10];
  const float* b_fc2    = (const float*)d_in[11];
  float* out = (float*)d_out;

  const int N = in_sizes[0] / HID;   // 20000
  const int E = in_sizes[1] / 2;     // 640000
  const int G = out_size / OUTD;     // 32
  const int NH = N * HID;

  const int* row = ei;
  const int* col = ei + E;

  char* ws = (char*)d_ws;
  size_t off = 0;
  auto carve = [&](size_t bytes) -> void* {
    void* p = ws + off;
    off = (off + bytes + 255) & ~(size_t)255;
    return p;
  };
  float*    h      = (float*)   carve((size_t)NH * 4);  // gemm1 out -> later h_sage
  float*    agg    = (float*)   carve((size_t)NH * 4);  // gcn accum -> h_gcn
  float*    neigh  = (float*)   carve((size_t)NH * 4);
  _Float16* actA   = (_Float16*)carve((size_t)NH * 2);  // xh, later neigh_h
  _Float16* actB   = (_Float16*)carve((size_t)NH * 2);  // h_gcn f16
  float*    deg    = (float*)   carve((size_t)N * 4);
  float*    dinv   = (float*)   carve((size_t)N * 4);
  float*    cnt    = (float*)   carve((size_t)N * 4);
  float*    pooled = (float*)   carve((size_t)G * HID * 4);
  float*    z      = (float*)   carve((size_t)G * HID * 4);
  _Float16* pooledh= (_Float16*)carve((size_t)G * HID * 2);
  _Float16* WtG    = (_Float16*)carve((size_t)HID * HID * 2);
  _Float16* WtL    = (_Float16*)carve((size_t)HID * HID * 2);
  _Float16* WtR    = (_Float16*)carve((size_t)HID * HID * 2);
  _Float16* Wt1    = (_Float16*)carve((size_t)HID * HID * 2);

  const int TB = 256;
  const dim3 blk(TB);
  const int wavesPerBlk = TB / 32;
  const int wGrid = (HID * HID + TB - 1) / TB;

  // 0) one-shot weight transpose+convert, activation convert
  wt_convert<<<dim3(wGrid), blk, 0, stream>>>(W_gcn,    WtG);
  wt_convert<<<dim3(wGrid), blk, 0, stream>>>(W_sage_l, WtL);
  wt_convert<<<dim3(wGrid), blk, 0, stream>>>(W_sage_r, WtR);
  wt_convert<<<dim3(wGrid), blk, 0, stream>>>(W_fc1,    Wt1);
  f32_to_f16<<<dim3((NH + TB - 1) / TB), blk, 0, stream>>>(x, actA, NH);

  // 1) h = x @ W_gcn            (TDM-staged WMMA)
  gemm128_f16<<<dim3(N / 16), blk, 0, stream>>>(actA, WtG, nullptr, nullptr,
                                                nullptr, h, 0);
  // 2) degrees (self loop contributes 1)
  fill_f32<<<dim3((N + TB - 1) / TB), blk, 0, stream>>>(deg, N, 1.0f);
  deg_scatter<<<dim3((E + TB - 1) / TB), blk, 0, stream>>>(col, deg, E);
  dinv_kernel<<<dim3((N + TB - 1) / TB), blk, 0, stream>>>(deg, dinv, N);
  // 3) GCN aggregation + bias + relu
  agg_init<<<dim3((NH + TB - 1) / TB), blk, 0, stream>>>(h, dinv, agg, NH);
  gcn_edge_scatter<<<dim3((E + wavesPerBlk - 1) / wavesPerBlk), blk, 0, stream>>>(
      h, dinv, row, col, agg, E);
  bias_relu_inplace<<<dim3((NH + TB - 1) / TB), blk, 0, stream>>>(agg, b_gcn, NH);
  // agg now holds h_gcn
  // 4) SAGE mean aggregation
  fill_f32<<<dim3((N + TB - 1) / TB), blk, 0, stream>>>(cnt, N, 0.0f);
  fill_f32<<<dim3((NH + TB - 1) / TB), blk, 0, stream>>>(neigh, NH, 0.0f);
  sage_edge_scatter<<<dim3((E + wavesPerBlk - 1) / wavesPerBlk), blk, 0, stream>>>(
      agg, row, col, neigh, cnt, E);
  sage_norm<<<dim3((NH + TB - 1) / TB), blk, 0, stream>>>(neigh, cnt, NH);
  // 5) h_sage = relu(neigh @ Wl + h_gcn @ Wr + b)   (dual WMMA, reuse h)
  f32_to_f16<<<dim3((NH + TB - 1) / TB), blk, 0, stream>>>(neigh, actA, NH);
  f32_to_f16<<<dim3((NH + TB - 1) / TB), blk, 0, stream>>>(agg,   actB, NH);
  gemm128_f16<<<dim3(N / 16), blk, 0, stream>>>(actA, WtL, actB, WtR,
                                                b_sage, h, 1);
  // 6) global add pool
  fill_f32<<<dim3((G * HID + TB - 1) / TB), blk, 0, stream>>>(pooled, G * HID, 0.0f);
  pool_scatter<<<dim3((NH + TB - 1) / TB), blk, 0, stream>>>(h, batch, pooled, NH);
  // 7) MLP head
  f32_to_f16<<<dim3((G * HID + TB - 1) / TB), blk, 0, stream>>>(pooled, pooledh, G * HID);
  gemm128_f16<<<dim3(G / 16), blk, 0, stream>>>(pooledh, Wt1, nullptr, nullptr,
                                                b_fc1, z, 1);
  fc2_kernel<<<dim3(1), dim3(128), 0, stream>>>(z, W_fc2, b_fc2, out, G);
}
